// WanSelfAttention_86998857547913
// MI455X (gfx1250) — matrix operand order repaired and checked by reading, not compiled, not run on hardware
//
#include <hip/hip_runtime.h>
#include <hip/hip_bf16.h>

// ---------------- problem constants (match reference) ----------------
#define DIMC   1536
#define HEADS  12
#define HD     128
#define SEQ    4096          // F*H*W = 4*32*32
#define CHALF  64            // HD/2
#define C1B    22            // CHALF - 2*(CHALF/3)
#define C2B    21            // CHALF/3
#define EPSF   1e-6f

typedef __attribute__((ext_vector_type(16))) _Float16 v16h;
typedef __attribute__((ext_vector_type(8)))  _Float16 v8h;
typedef __attribute__((ext_vector_type(8)))  float    v8f;
typedef int v4i_vec __attribute__((vector_size(16)));   // matches builtin param

// ---------------- async global->LDS copy (CDNA5 ASYNCcnt path), guarded ----
#if __has_builtin(__builtin_amdgcn_global_load_async_to_lds_b128)
#define HAVE_ASYNC_LDS 1
#else
#define HAVE_ASYNC_LDS 0
#endif

// copy 16 bytes global -> LDS per lane
static __device__ __forceinline__ void cp16(_Float16* lds_dst,
                                            const _Float16* gsrc) {
#if HAVE_ASYNC_LDS
  // signature (from hipcc diagnostic): (v4i* gsrc, v4i* lds, imm offset, imm cpol)
  __builtin_amdgcn_global_load_async_to_lds_b128((v4i_vec*)gsrc,
                                                 (v4i_vec*)lds_dst, 0, 0);
#else
  *(v8h*)lds_dst = *(const v8h*)gsrc;
#endif
}

static __device__ __forceinline__ void wait_async0() {
#if HAVE_ASYNC_LDS
#if __has_builtin(__builtin_amdgcn_s_wait_asynccnt)
  __builtin_amdgcn_s_wait_asynccnt(0);
#else
  asm volatile("s_wait_asynccnt 0x0" ::: "memory");
#endif
#endif
}
static __device__ __forceinline__ void wait_async2() {
#if HAVE_ASYNC_LDS
#if __has_builtin(__builtin_amdgcn_s_wait_asynccnt)
  __builtin_amdgcn_s_wait_asynccnt(2);
#else
  asm volatile("s_wait_asynccnt 0x2" ::: "memory");
#endif
#endif
}

// ---------------- WMMA fragment helpers (wave32, 16x16x32 f16) ----------------
// A fragment (16x32, 16-bit): lanes 0-15 -> rows M, halves K=[0..7,16..23];
// lanes 16-31 -> rows M, halves K=[8..15,24..31].  (ISA 7.12.2)
static __device__ __forceinline__ v16h load_a_frag(const _Float16* base, int ld,
                                                   int row0, int k0) {
  const int lane = threadIdx.x & 31;
  const int r    = row0 + (lane & 15);
  const int kb   = (lane < 16) ? 0 : 8;
  const _Float16* p = base + (size_t)r * ld + k0 + kb;
  v8h lo = *(const v8h*)(p);        // K = kb .. kb+7
  v8h hi = *(const v8h*)(p + 16);   // K = kb+16 .. kb+23
  return __builtin_shufflevector(lo, hi, 0,1,2,3,4,5,6,7,8,9,10,11,12,13,14,15);
}

// B fragment (32x16, 16-bit): lane = column N (mod 16), halves = K [0..7,16..23]
// for lanes 0-15, K [8..15,24..31] for lanes 16-31.  Source stored as [N][K] row-major.
static __device__ __forceinline__ v16h load_b_frag(const _Float16* base, int ld,
                                                   int n0, int k0) {
  const int lane = threadIdx.x & 31;
  const _Float16* p = base + (size_t)(n0 + (lane & 15)) * ld
                           + k0 + ((lane < 16) ? 0 : 8);
  v8h lo = *(const v8h*)(p);
  v8h hi = *(const v8h*)(p + 16);
  return __builtin_shufflevector(lo, hi, 0,1,2,3,4,5,6,7,8,9,10,11,12,13,14,15);
}

static __device__ __forceinline__ v8f wmma_f16(v16h a, v16h b, v8f c) {
  return __builtin_amdgcn_wmma_f32_16x16x32_f16(false, a, false, b, (short)0, c,
                                                false, false);
}

// ---------------- conversion / transpose ----------------
__global__ void cvt_f32_to_f16_kernel(const float* __restrict__ in,
                                      _Float16* __restrict__ out, int n) {
  for (int i = blockIdx.x * blockDim.x + threadIdx.x; i < n;
       i += gridDim.x * blockDim.x)
    out[i] = (_Float16)in[i];
}

// WT[n*DIM + k] = (f16) W[k*DIM + n]
__global__ __launch_bounds__(256) void transpose_to_f16_kernel(
    const float* __restrict__ W, _Float16* __restrict__ WT) {
  __shared__ float tile[16][17];
  const int tx = threadIdx.x & 15, ty = threadIdx.x >> 4;
  const int n0 = blockIdx.x * 16, k0 = blockIdx.y * 16;
  tile[ty][tx] = W[(size_t)(k0 + ty) * DIMC + n0 + tx];
  __syncthreads();
  WT[(size_t)(n0 + ty) * DIMC + k0 + tx] = (_Float16)tile[tx][ty];
}

// ---------------- WMMA GEMM: out = A(f16) * BT(f16)^T + bias ----------------
// A: [SEQ, DIMC] f16, BT: [DIMC(out), DIMC(in)] f16, out: f32 [SEQ, DIMC].
// block = 256 (8 waves); block tile 128x128; wave w -> 16x128 stripe.
// B panel (128 x 32k = 8KB) double-buffered in LDS, async-staged when available.
__global__ __launch_bounds__(256) void gemm16_bias_kernel(
    const _Float16* __restrict__ A, const _Float16* __restrict__ BT,
    const float* __restrict__ bias, float* __restrict__ out) {
  __shared__ __align__(16) _Float16 bstage[2][128 * 32];  // [n_local][k 0..31]
  const int tid  = threadIdx.x;
  const int wave = tid >> 5, lane = tid & 31;
  const int m0 = blockIdx.y * 128 + wave * 16;
  const int n0 = blockIdx.x * 128;
  const int snl  = tid >> 2;         // staging row 0..63 (+64)
  const int sseg = (tid & 3) * 8;    // staging half-offset 0,8,16,24

  v8f acc[8] = {};

  auto stage = [&](int b, int k0) {
#pragma unroll
    for (int i = 0; i < 2; ++i) {
      const int nl = snl + i * 64;
      cp16(&bstage[b][nl * 32 + sseg],
           BT + (size_t)(n0 + nl) * DIMC + k0 + sseg);
    }
  };

  const int NK = DIMC / 32;
  stage(0, 0);
  for (int ks = 0; ks < NK; ++ks) {
    const int cur = ks & 1;
    if (ks + 1 < NK) {
      stage(cur ^ 1, (ks + 1) * 32);       // overlap next panel with compute
      wait_async2();                        // first 2 (cur panel) complete
    } else {
      wait_async0();
    }
    if (ks + 2 < NK)                        // pull A ahead into the WGP
      __builtin_prefetch(A + (size_t)(m0 + (lane & 15)) * DIMC + (ks + 2) * 32,
                         0, 3);
    __syncthreads();                        // cur panel visible to all waves
    v16h a = load_a_frag(A, DIMC, m0, ks * 32);
    const _Float16* bp = &bstage[cur][0];
#pragma unroll
    for (int t = 0; t < 8; ++t) {
      v16h b = load_b_frag(bp, 32, t * 16, 0);
      acc[t] = wmma_f16(a, b, acc[t]);
    }
    __syncthreads();                        // reads done before panel reuse
  }

  const int col = lane & 15;
  const int rb  = (lane < 16) ? 0 : 8;
#pragma unroll
  for (int t = 0; t < 8; ++t) {
    const float bn = bias[n0 + t * 16 + col];
#pragma unroll
    for (int v = 0; v < 8; ++v)
      out[(size_t)(m0 + rb + v) * DIMC + n0 + t * 16 + col] = acc[t][v] + bn;
  }
}

// ---------------- RMSNorm + RoPE + V transpose ----------------
__global__ __launch_bounds__(256) void rms_rope_kernel(
    const float* __restrict__ Qf, const float* __restrict__ Kf,
    const float* __restrict__ Vf, const float* __restrict__ g_q,
    const float* __restrict__ g_k, const float* __restrict__ freqs,
    _Float16* __restrict__ Qh, _Float16* __restrict__ Kh,
    _Float16* __restrict__ Vth) {
  const int s = blockIdx.x;
  const int tid = threadIdx.x;
  const float* qrow = Qf + (size_t)s * DIMC;
  const float* krow = Kf + (size_t)s * DIMC;
  const float* vrow = Vf + (size_t)s * DIMC;

  float sq = 0.f, sk = 0.f;
  for (int i = tid; i < DIMC; i += 256) {
    float q = qrow[i]; sq += q * q;
    float k = krow[i]; sk += k * k;
  }
#pragma unroll
  for (int off = 16; off; off >>= 1) {
    sq += __shfl_xor(sq, off, 32);
    sk += __shfl_xor(sk, off, 32);
  }
  __shared__ float red[2][8];
  const int wave = tid >> 5, lane = tid & 31;
  if (lane == 0) { red[0][wave] = sq; red[1][wave] = sk; }
  __syncthreads();
  sq = 0.f; sk = 0.f;
#pragma unroll
  for (int w = 0; w < 8; ++w) { sq += red[0][w]; sk += red[1][w]; }
  const float rq = rsqrtf(sq * (1.0f / DIMC) + EPSF);
  const float rk = rsqrtf(sk * (1.0f / DIMC) + EPSF);

  const int fidx = s >> 10;
  const int hidx = (s >> 5) & 31;
  const int widx = s & 31;

  for (int p = tid; p < HEADS * CHALF; p += 256) {
    const int head = p >> 6;
    const int pp   = p & 63;
    float ang;
    if (pp < C1B)            ang = freqs[fidx * CHALF + pp];
    else if (pp < C1B + C2B) ang = freqs[hidx * CHALF + pp];
    else                     ang = freqs[widx * CHALF + pp];
    const float c  = __cosf(ang);
    const float si = __sinf(ang);
    const int d = head * HD + 2 * pp;
    const size_t idx = (size_t)s * DIMC + d;
    const float q_r = qrow[d]     * rq * g_q[d];
    const float q_i = qrow[d + 1] * rq * g_q[d + 1];
    Qh[idx]     = (_Float16)(q_r * c - q_i * si);
    Qh[idx + 1] = (_Float16)(q_r * si + q_i * c);
    const float k_r = krow[d]     * rk * g_k[d];
    const float k_i = krow[d + 1] * rk * g_k[d + 1];
    Kh[idx]     = (_Float16)(k_r * c - k_i * si);
    Kh[idx + 1] = (_Float16)(k_r * si + k_i * c);
  }
  for (int i = tid; i < DIMC; i += 256)
    Vth[(size_t)i * SEQ + s] = (_Float16)vrow[i];
}

// ---------------- flash attention ----------------
// grid = (SEQ/128, HEADS), block = 256 (8 waves); wave = one 16-query tile.
// Per 32-key block: K panel (32x128) and V^T panel (128x32) staged once into
// LDS for all 8 waves; P tile re-shaped through per-wave LDS for the PV WMMA.
__global__ __launch_bounds__(256) void attn_fa_kernel(
    const _Float16* __restrict__ Qh, const _Float16* __restrict__ Kh,
    const _Float16* __restrict__ Vth, const int* __restrict__ seq_lens,
    _Float16* __restrict__ Oh) {
  __shared__ __align__(16) _Float16 kbuf[32 * 128];    // [key_local][hd]
  __shared__ __align__(16) _Float16 vbuf[128 * 32];    // [hd_local][key_local]
  __shared__ __align__(16) _Float16 pbuf[8][16 * 32];  // per-wave P tile
  const int tid  = threadIdx.x;
  const int wave = tid >> 5;
  const int lane = tid & 31;
  const int head = blockIdx.y;
  const int q0   = (blockIdx.x * 8 + wave) * 16;
  const int col  = lane & 15;
  const int rb   = (lane < 16) ? 0 : 8;
  const int seqlen = seq_lens[0];
  const float scale = 0.08838834764831845f;  // 1/sqrt(128)

  // staging coords
  const int krow_s = tid >> 4;          // 0..15 (+16)
  const int kseg_s = (tid & 15) * 8;    // 0..120 halves
  const int vrow_s = tid >> 2;          // 0..63 (+64)
  const int vseg_s = (tid & 3) * 8;     // 0,8,16,24 halves

  v16h aq[4];
#pragma unroll
  for (int c = 0; c < 4; ++c)
    aq[c] = load_a_frag(Qh, DIMC, q0, head * HD + c * 32);

  v8f o[8] = {};
  float mrow[8], lrow[8];
#pragma unroll
  for (int v = 0; v < 8; ++v) { mrow[v] = -1e30f; lrow[v] = 0.f; }

  _Float16* pw = &pbuf[wave][0];

  for (int key0 = 0; key0 < SEQ; key0 += 32) {
    // ---- stage K / V^T panels for this 32-key block ----
#pragma unroll
    for (int i = 0; i < 2; ++i) {
      const int kr = krow_s + i * 16;
      cp16(&kbuf[kr * 128 + kseg_s],
           Kh + (size_t)(key0 + kr) * DIMC + head * HD + kseg_s);
      const int vr = vrow_s + i * 64;
      cp16(&vbuf[vr * 32 + vseg_s],
           Vth + (size_t)(head * HD + vr) * SEQ + key0 + vseg_s);
    }
    if (key0 + 32 < SEQ) {  // pull next panels toward the WGP
      __builtin_prefetch(Kh + (size_t)(key0 + 32 + krow_s) * DIMC + head * HD,
                         0, 3);
      __builtin_prefetch(Vth + (size_t)(head * HD + vrow_s) * SEQ + key0 + 32,
                         0, 3);
    }
    wait_async0();
    __syncthreads();  // panels visible

    // ---- scores: 16 queries x 32 keys ----
    v8f st0 = {}, st1 = {};
#pragma unroll
    for (int c = 0; c < 4; ++c) {
      st0 = wmma_f16(aq[c], load_b_frag(kbuf, 128, 0,  c * 32), st0);
      st1 = wmma_f16(aq[c], load_b_frag(kbuf, 128, 16, c * 32), st1);
    }
    const int c0 = key0 + col, c1 = c0 + 16;
    // ---- online softmax per row (row's cols live in one 16-lane half) ----
#pragma unroll
    for (int v = 0; v < 8; ++v) {
      float s0 = st0[v] * scale; if (c0 >= seqlen) s0 = -1e9f;
      float s1 = st1[v] * scale; if (c1 >= seqlen) s1 = -1e9f;
      float mx = fmaxf(s0, s1);
      mx = fmaxf(mx, __shfl_xor(mx, 1, 16));
      mx = fmaxf(mx, __shfl_xor(mx, 2, 16));
      mx = fmaxf(mx, __shfl_xor(mx, 4, 16));
      mx = fmaxf(mx, __shfl_xor(mx, 8, 16));
      const float mnew  = fmaxf(mrow[v], mx);
      const float alpha = __expf(mrow[v] - mnew);
      const float p0 = __expf(s0 - mnew);
      const float p1 = __expf(s1 - mnew);
      float ps = p0 + p1;
      ps += __shfl_xor(ps, 1, 16);
      ps += __shfl_xor(ps, 2, 16);
      ps += __shfl_xor(ps, 4, 16);
      ps += __shfl_xor(ps, 8, 16);
      lrow[v] = lrow[v] * alpha + ps;
      mrow[v] = mnew;
      const int r = rb + v;
      pw[r * 32 + col]      = (_Float16)p0;
      pw[r * 32 + col + 16] = (_Float16)p1;
#pragma unroll
      for (int t = 0; t < 8; ++t) o[t][v] *= alpha;
    }
    __syncthreads();  // P visible; K-panel reads done

    // ---- O += P * V ----
    const v16h ap = load_a_frag(pw, 32, 0, 0);
#pragma unroll
    for (int t = 0; t < 8; ++t) {
      v16h bv = load_b_frag(vbuf, 32, t * 16, 0);
      o[t] = wmma_f16(ap, bv, o[t]);
    }
    __syncthreads();  // all panel reads done before restage
  }

#pragma unroll
  for (int v = 0; v < 8; ++v) {
    const float inv = 1.0f / lrow[v];
    const int r = q0 + rb + v;
#pragma unroll
    for (int t = 0; t < 8; ++t)
      Oh[(size_t)r * DIMC + head * HD + t * 16 + col] =
          (_Float16)(o[t][v] * inv);
  }
}

// ---------------- launch ----------------
extern "C" void kernel_launch(void* const* d_in, const int* in_sizes, int n_in,
                              void* d_out, int out_size, void* d_ws, size_t ws_size,
                              hipStream_t stream) {
  (void)in_sizes; (void)n_in; (void)out_size; (void)ws_size;
  const float* x        = (const float*)d_in[0];
  const int*   seq_lens = (const int*)d_in[1];
  // d_in[2] = grid_sizes (4,32,32) folded into compile-time constants
  const float* freqs    = (const float*)d_in[3];
  const float* Wq = (const float*)d_in[4];
  const float* bq = (const float*)d_in[5];
  const float* Wk = (const float*)d_in[6];
  const float* bk = (const float*)d_in[7];
  const float* Wv = (const float*)d_in[8];
  const float* bv = (const float*)d_in[9];
  const float* Wo = (const float*)d_in[10];
  const float* bo = (const float*)d_in[11];
  const float* g_q = (const float*)d_in[12];
  const float* g_k = (const float*)d_in[13];

  char* ws = (char*)d_ws;
  size_t off = 0;
  _Float16* xh  = (_Float16*)(ws + off); off += (size_t)SEQ * DIMC * 2;
  _Float16* WqT = (_Float16*)(ws + off); off += (size_t)DIMC * DIMC * 2;
  _Float16* WkT = (_Float16*)(ws + off); off += (size_t)DIMC * DIMC * 2;
  _Float16* WvT = (_Float16*)(ws + off); off += (size_t)DIMC * DIMC * 2;
  _Float16* WoT = (_Float16*)(ws + off); off += (size_t)DIMC * DIMC * 2;
  float*    Qf  = (float*)(ws + off);    off += (size_t)SEQ * DIMC * 4;
  float*    Kf  = (float*)(ws + off);    off += (size_t)SEQ * DIMC * 4;
  float*    Vf  = (float*)(ws + off);    off += (size_t)SEQ * DIMC * 4;
  _Float16* Qh  = (_Float16*)(ws + off); off += (size_t)SEQ * DIMC * 2;
  _Float16* Kh  = (_Float16*)(ws + off); off += (size_t)SEQ * DIMC * 2;
  _Float16* Vth = (_Float16*)(ws + off); off += (size_t)DIMC * SEQ * 2;
  _Float16* Oh  = (_Float16*)(ws + off); off += (size_t)SEQ * DIMC * 2;

  const dim3 blk(256);

  cvt_f32_to_f16_kernel<<<2048, blk, 0, stream>>>(x, xh, SEQ * DIMC);

  const dim3 tgrid(DIMC / 16, DIMC / 16);
  transpose_to_f16_kernel<<<tgrid, blk, 0, stream>>>(Wq, WqT);
  transpose_to_f16_kernel<<<tgrid, blk, 0, stream>>>(Wk, WkT);
  transpose_to_f16_kernel<<<tgrid, blk, 0, stream>>>(Wv, WvT);
  transpose_to_f16_kernel<<<tgrid, blk, 0, stream>>>(Wo, WoT);

  const dim3 ggrid(DIMC / 128, SEQ / 128);
  gemm16_bias_kernel<<<ggrid, blk, 0, stream>>>(xh, WqT, bq, Qf);
  gemm16_bias_kernel<<<ggrid, blk, 0, stream>>>(xh, WkT, bk, Kf);
  gemm16_bias_kernel<<<ggrid, blk, 0, stream>>>(xh, WvT, bv, Vf);

  rms_rope_kernel<<<SEQ, blk, 0, stream>>>(Qf, Kf, Vf, g_q, g_k, freqs,
                                           Qh, Kh, Vth);

  const dim3 agrid(SEQ / 128, HEADS);
  attn_fa_kernel<<<agrid, blk, 0, stream>>>(Qh, Kh, Vth, seq_lens, Oh);

  gemm16_bias_kernel<<<ggrid, blk, 0, stream>>>(Oh, WoT, bo, (float*)d_out);
}
